// BaseDependentAttentionLayer_55198919688627
// MI455X (gfx1250) — compile-verified, matrix operand order checked by reading, not run on hardware
//
#include <hip/hip_runtime.h>
#include <hip/hip_bf16.h>

typedef __attribute__((ext_vector_type(16))) _Float16 v16h;
typedef __attribute__((ext_vector_type(8)))  _Float16 v8h;
typedef __attribute__((ext_vector_type(4)))  _Float16 v4h;
typedef __attribute__((ext_vector_type(8)))  float    v8f;

#define N_NODES 50000
#define N_EDGES 800000
#define DIM 128
#define HEADS 8
#define HEAD_DIM 16

// ---------------------------------------------------------------- atomics
__device__ inline void atomicAddF(float* p, float v) {
    __hip_atomic_fetch_add(p, v, __ATOMIC_RELAXED, __HIP_MEMORY_SCOPE_AGENT);
}
// IEEE-ordered float max via signed/unsigned integer atomics (init = -inf bits)
__device__ inline void atomicMaxF(int* p, float v) {
    if (v >= 0.0f) atomicMax(p, __float_as_int(v));
    else           atomicMin((unsigned int*)p, __float_as_uint(v));
}

// ---------------------------------------------------------------- WMMA tile
// One wave computes a 16x16 f32 tile of (A[16x128] @ W[128x128])[:, n0:n0+16].
// A fragments preloaded (f16). WT is the weight matrix TRANSPOSED (N-major,
// row n = output column n, K contiguous) so each lane's B fragment is two
// contiguous 16-byte runs -> two global_load_b128 per K-step, no converts.
__device__ inline void gemm_tile_16x16(const v16h a[4], const _Float16* __restrict__ WT,
                                       const float* __restrict__ bias,
                                       float* __restrict__ out,
                                       int r0, int n0, int lane) {
    const int col = lane & 15;
    const int kb  = (lane >> 4) << 3;      // 0 or 8
    const _Float16* wr = WT + (size_t)(n0 + col) * DIM;
    v8f c = {};
#pragma unroll
    for (int t = 0; t < 4; ++t) {
        // element i <-> k = t*32 + kb + i + (i>=8 ? 8 : 0)
        const v8h lo = *(const v8h*)(wr + t * 32 + kb);
        const v8h hi = *(const v8h*)(wr + t * 32 + kb + 16);
        v16h b;
#pragma unroll
        for (int i = 0; i < 8; ++i) { b[i] = lo[i]; b[i + 8] = hi[i]; }
        c = __builtin_amdgcn_wmma_f32_16x16x32_f16(false, a[t], false, b,
                                                   (short)0, c, false, false);
    }
    const float bia = bias[n0 + col];
    float* op = out + (size_t)r0 * DIM + n0 + col;
#pragma unroll
    for (int j = 0; j < 8; ++j) {
        int m = j + ((lane >> 4) << 3);
        op[(size_t)m * DIM] = c[j] + bia;
    }
}

// Load four 16x32 f16 A-fragments for rows [r0, r0+16) from row-major f16 src.
__device__ inline void load_a_frags(const _Float16* __restrict__ src, int r0, int lane,
                                    v16h a[4]) {
    const int row = lane & 15;
    const int kb  = (lane >> 4) << 3;      // 0 or 8
    const _Float16* xr = src + (size_t)(r0 + row) * DIM;
#pragma unroll
    for (int t = 0; t < 4; ++t) {
        const v8h lo = *(const v8h*)(xr + t * 32 + kb);
        const v8h hi = *(const v8h*)(xr + t * 32 + kb + 16);
#pragma unroll
        for (int i = 0; i < 8; ++i) { a[t][i] = lo[i]; a[t][i + 8] = hi[i]; }
    }
}

// ---------------------------------------------------------------- kernels
// 1) init softmax state (must run every call: graph replay does not re-poison)
__global__ void init_state(int* segmax_bits, float* denom, float* vals) {
    int i = blockIdx.x * blockDim.x + threadIdx.x;
    if (i < N_NODES * HEADS) {
        segmax_bits[i] = 0xFF800000;       // -inf
        denom[i] = 0.0f;
    }
    if (i < N_NODES * DIM) vals[i] = 0.0f;
}

// 1b) f32 -> f16 elementwise (4 elems/thread)
__global__ void cvt_f32_f16_x4(const float* __restrict__ src, _Float16* __restrict__ dst) {
    int i = blockIdx.x * blockDim.x + threadIdx.x;
    float4 vv = ((const float4*)src)[i];
    v4h o = { (_Float16)vv.x, (_Float16)vv.y, (_Float16)vv.z, (_Float16)vv.w };
    ((v4h*)dst)[i] = o;
}

// 1c) convert + transpose the four 128x128 weight matrices into f16, N-major:
//     WT[m][n*DIM + k] = (f16) W_m[k*DIM + n]
__global__ void cvt_weightsT(const float* __restrict__ Wq, const float* __restrict__ Wk,
                             const float* __restrict__ Wv, const float* __restrict__ Wo,
                             _Float16* __restrict__ WT) {
    int i = blockIdx.x * blockDim.x + threadIdx.x;   // 4*DIM*DIM threads
    int m = i >> 14, r = i & 16383, n = r >> 7, k = r & 127;
    const float* W = (m == 0) ? Wq : (m == 1) ? Wk : (m == 2) ? Wv : Wo;
    WT[i] = (_Float16)W[(size_t)k * DIM + n];
}

// 2) fused QKV projection: 8 waves/block, wave w owns column tile w*16; one
//    M-tile (16 node rows) per block; A fragments reused across Q/K/V.
__global__ void qkv_gemm(const _Float16* __restrict__ x16,
                         const _Float16* __restrict__ WTq, const float* __restrict__ bq,
                         const _Float16* __restrict__ WTk, const float* __restrict__ bk,
                         const _Float16* __restrict__ WTv, const float* __restrict__ bv,
                         float* __restrict__ q, float* __restrict__ k,
                         float* __restrict__ v) {
    const int lane = threadIdx.x & 31;
    const int wave = threadIdx.x >> 5;
    const int r0 = blockIdx.x * 16;
    const int n0 = wave * 16;
    v16h a[4];
    load_a_frags(x16, r0, lane, a);
    gemm_tile_16x16(a, WTq, bq, q, r0, n0, lane);
    gemm_tile_16x16(a, WTk, bk, k, r0, n0, lane);
    gemm_tile_16x16(a, WTv, bv, v, r0, n0, lane);
}

// 3) per-(edge,head) score = scale * ew * <q[origin,h], k[dest,h]>, and segment max
__global__ void edge_scores(const int* __restrict__ origin, const int* __restrict__ dest,
                            const float* __restrict__ ew,
                            const float* __restrict__ q, const float* __restrict__ k,
                            float* __restrict__ scores, int* __restrict__ segmax_bits) {
    int idx = blockIdx.x * blockDim.x + threadIdx.x;      // E*H threads
    int e = idx >> 3, h = idx & 7;
    int o = origin[e], d = dest[e];
    const float4* qp = (const float4*)(q + (size_t)o * DIM + h * HEAD_DIM);
    const float4* kp = (const float4*)(k + (size_t)d * DIM + h * HEAD_DIM);
    float s = 0.0f;
#pragma unroll
    for (int t = 0; t < 4; ++t) {
        float4 av = qp[t], bv = kp[t];
        s += av.x * bv.x + av.y * bv.y + av.z * bv.z + av.w * bv.w;
    }
    s *= 0.25f;                 // HEAD_DIM^-0.5
    s *= ew[idx];               // ew is [E, H], idx == e*H + h
    scores[idx] = s;
    atomicMaxF(&segmax_bits[o * HEADS + h], s);
}

// 4) e = exp(score - segmax[origin]); accumulate denominators
__global__ void edge_exp(const int* __restrict__ origin,
                         const int* __restrict__ segmax_bits,
                         float* __restrict__ scores, float* __restrict__ denom) {
    int idx = blockIdx.x * blockDim.x + threadIdx.x;      // E*H threads
    int e = idx >> 3, h = idx & 7;
    int o = origin[e];
    float sm = __int_as_float(segmax_bits[o * HEADS + h]);
    if (!isfinite(sm)) sm = 0.0f;
    float ev = __expf(scores[idx] - sm);
    scores[idx] = ev;                                      // in-place: now holds exp
    atomicAddF(&denom[o * HEADS + h], ev);
}

// 5) scatter: vals[origin] += attn * v[dest]  (thread per (edge, dim): coalesced)
__global__ void edge_scatter(const int* __restrict__ origin, const int* __restrict__ dest,
                             const float* __restrict__ ev, const float* __restrict__ denom,
                             const float* __restrict__ v, float* __restrict__ vals) {
    int idx = blockIdx.x * blockDim.x + threadIdx.x;      // E*DIM threads
    int e = idx >> 7, dim = idx & 127, h = dim >> 4;
    int o = origin[e], d = dest[e];
    float attn = ev[e * HEADS + h] / (denom[o * HEADS + h] + 1e-16f);
    atomicAddF(&vals[(size_t)o * DIM + dim], attn * v[(size_t)d * DIM + dim]);
}

// 6) output projection: out = vals @ Wo + bo
__global__ void out_gemm(const _Float16* __restrict__ vals16,
                         const _Float16* __restrict__ WTo, const float* __restrict__ bo,
                         float* __restrict__ out) {
    const int lane = threadIdx.x & 31;
    const int wave = threadIdx.x >> 5;
    const int r0 = blockIdx.x * 16;
    const int n0 = wave * 16;
    v16h a[4];
    load_a_frags(vals16, r0, lane, a);
    gemm_tile_16x16(a, WTo, bo, out, r0, n0, lane);
}

// 7) LayerNorm + residual: one wave32 per node, 4 dims per lane
__global__ void ln_residual(const float* __restrict__ outp, const float* __restrict__ x,
                            const float* __restrict__ gamma, const float* __restrict__ beta,
                            float* __restrict__ y) {
    const int lane = threadIdx.x & 31;
    const int wave = threadIdx.x >> 5;
    const int node = blockIdx.x * 8 + wave;
    const float4 ov = ((const float4*)(outp + (size_t)node * DIM))[lane];
    float s = ov.x + ov.y + ov.z + ov.w;
#pragma unroll
    for (int m = 16; m >= 1; m >>= 1) s += __shfl_xor(s, m, 32);
    const float mu = s * (1.0f / DIM);
    float dx0 = ov.x - mu, dx1 = ov.y - mu, dx2 = ov.z - mu, dx3 = ov.w - mu;
    float s2 = dx0 * dx0 + dx1 * dx1 + dx2 * dx2 + dx3 * dx3;
#pragma unroll
    for (int m = 16; m >= 1; m >>= 1) s2 += __shfl_xor(s2, m, 32);
    const float inv = rsqrtf(s2 * (1.0f / DIM) + 1e-5f);
    const float4 xv = ((const float4*)(x + (size_t)node * DIM))[lane];
    const float4 gv = ((const float4*)gamma)[lane];
    const float4 bv = ((const float4*)beta)[lane];
    float4 r;
    r.x = xv.x + dx0 * inv * gv.x + bv.x;
    r.y = xv.y + dx1 * inv * gv.y + bv.y;
    r.z = xv.z + dx2 * inv * gv.z + bv.z;
    r.w = xv.w + dx3 * inv * gv.w + bv.w;
    ((float4*)(y + (size_t)node * DIM))[lane] = r;
}

// ---------------------------------------------------------------- launch
extern "C" void kernel_launch(void* const* d_in, const int* in_sizes, int n_in,
                              void* d_out, int out_size, void* d_ws, size_t ws_size,
                              hipStream_t stream) {
    const float* x   = (const float*)d_in[0];
    const int*   ei  = (const int*)  d_in[1];   // [2, E]
    const float* ew  = (const float*)d_in[2];   // [E, H]
    const float* Wq  = (const float*)d_in[3];
    const float* bq  = (const float*)d_in[4];
    const float* Wk  = (const float*)d_in[5];
    const float* bk  = (const float*)d_in[6];
    const float* Wv  = (const float*)d_in[7];
    const float* bv  = (const float*)d_in[8];
    const float* Wo  = (const float*)d_in[9];
    const float* bo  = (const float*)d_in[10];
    const float* gam = (const float*)d_in[11];
    const float* bet = (const float*)d_in[12];
    float* y = (float*)d_out;

    const int* origin = ei;
    const int* dest   = ei + N_EDGES;

    // workspace layout
    float* ws   = (float*)d_ws;
    float* q    = ws;                               // N*DIM f32
    float* k    = q    + (size_t)N_NODES * DIM;     // N*DIM f32
    float* v    = k    + (size_t)N_NODES * DIM;     // N*DIM f32
    float* sc   = v    + (size_t)N_NODES * DIM;     // E*H   f32 (scores, then exp)
    float* smx  = sc   + (size_t)N_EDGES * HEADS;   // N*H   (max bits)
    float* den  = smx  + (size_t)N_NODES * HEADS;   // N*H   f32
    float* vals = den  + (size_t)N_NODES * HEADS;   // N*DIM f32
    float* outb = vals + (size_t)N_NODES * DIM;     // N*DIM f32
    _Float16* h16    = (_Float16*)(outb + (size_t)N_NODES * DIM);
    _Float16* x16    = h16;                          // N*DIM f16
    _Float16* vals16 = x16 + (size_t)N_NODES * DIM;  // N*DIM f16
    _Float16* WT     = vals16 + (size_t)N_NODES * DIM; // 4*DIM*DIM f16
    _Float16* WTq = WT;
    _Float16* WTk = WT + 1 * DIM * DIM;
    _Float16* WTv = WT + 2 * DIM * DIM;
    _Float16* WTo = WT + 3 * DIM * DIM;

    init_state<<<(N_NODES * DIM + 255) / 256, 256, 0, stream>>>((int*)smx, den, vals);

    cvt_f32_f16_x4<<<(N_NODES * DIM / 4) / 256, 256, 0, stream>>>(x, x16);
    cvt_weightsT<<<(4 * DIM * DIM) / 256, 256, 0, stream>>>(Wq, Wk, Wv, Wo, WT);

    qkv_gemm<<<N_NODES / 16, 256, 0, stream>>>(x16, WTq, bq, WTk, bk, WTv, bv, q, k, v);

    edge_scores<<<(N_EDGES * HEADS) / 256, 256, 0, stream>>>(origin, dest, ew, q, k,
                                                             sc, (int*)smx);

    edge_exp<<<(N_EDGES * HEADS) / 256, 256, 0, stream>>>(origin, (const int*)smx,
                                                          sc, den);

    edge_scatter<<<(N_EDGES * DIM) / 256, 256, 0, stream>>>(origin, dest, sc, den,
                                                            v, vals);

    cvt_f32_f16_x4<<<(N_NODES * DIM / 4) / 256, 256, 0, stream>>>(vals, vals16);

    out_gemm<<<N_NODES / 16, 256, 0, stream>>>(vals16, WTo, bo, outb);

    ln_residual<<<N_NODES / 8, 256, 0, stream>>>(outb, x, gam, bet, y);
}
